// WindowAttention_65807488909317
// MI455X (gfx1250) — matrix operand order repaired and checked by reading, not compiled, not run on hardware
//
#include <hip/hip_runtime.h>
#include <hip/hip_bf16.h>

typedef __attribute__((ext_vector_type(16))) _Float16 v16h;
typedef __attribute__((ext_vector_type(8)))  float    v8f;

#define B__   64
#define H__   4
#define N__   625
#define NPAD  640
#define D__   32
#define C__   128
#define NW__  32

// workspace byte offsets
#define QOFF   0UL
#define KOFF   10485760UL      // 64*4*640*32*2
#define VTOFF  20971520UL
#define ZOFF   31457280UL      // Z: 40000*128*2  = 10,240,000
#define BOFF   41697280UL      // biasHNN: 4*625*625*4 = 6,250,000
#define XHOFF  47947280UL      // x in f16, padded: 64*640*128*2 = 10,485,760
#define WQOFF  58433040UL      // qkv_w f16: 384*128*2 = 98,304
#define WPOFF  58531344UL      // proj_w f16: 128*128*2 = 32,768
#define WSTOT  58564112UL
#define XHBYTES 10485760UL

union AF { v16h v; _Float16 h[16]; unsigned u[8]; };

static __device__ __forceinline__ int kbase_of(int j, int laneHalf) {
  // K index of element pair 2j within a 16-bit A-fragment (16x32), contiguous order k = 16*laneHalf + 2j
  return laneHalf * 16 + j * 2;
}

// ---------------- Kernel 0a: f32 -> f16 bulk convert (4 elems/thread), dense
__global__ __launch_bounds__(256) void cvt4_kernel(
    const float* __restrict__ src, _Float16* __restrict__ dst, int n4)
{
  int i = blockIdx.x * 256 + threadIdx.x;
  if (i >= n4) return;
  const float4 f = ((const float4*)src)[i];
  union { _Float16 h[4]; unsigned long long u; } pk;
  pk.h[0] = (_Float16)f.x; pk.h[1] = (_Float16)f.y;
  pk.h[2] = (_Float16)f.z; pk.h[3] = (_Float16)f.w;
  ((unsigned long long*)dst)[i] = pk.u;
}

// ---------------- Kernel 0b: x (64,625,128) f32 -> xh (64,640,128) f16 scatter
__global__ __launch_bounds__(256) void cvtx_kernel(
    const float* __restrict__ src, _Float16* __restrict__ dst)
{
  const int b = blockIdx.y;
  const int i = blockIdx.x * 256 + threadIdx.x;    // 4-elem chunk within batch b
  if (i >= N__ * (C__ / 4)) return;
  const int n  = i >> 5;                            // 128/4 = 32 chunks per row
  const int c4 = i & 31;
  const float4 f = ((const float4*)src)[((size_t)b * N__ + n) * 32 + c4];
  union { _Float16 h[4]; unsigned long long u; } pk;
  pk.h[0] = (_Float16)f.x; pk.h[1] = (_Float16)f.y;
  pk.h[2] = (_Float16)f.z; pk.h[3] = (_Float16)f.w;
  ((unsigned long long*)dst)[((size_t)b * NPAD + n) * 32 + c4] = pk.u;
}

// ---------------- Kernel 1: qkv = x @ qkv_w.T + qkv_b ; scatter to Qh/Kh/Vt (f16)
// grid = (40 n-tiles, 64 batches); no divisions, tile never straddles a batch
__global__ __launch_bounds__(256) void qkv_kernel(
    const _Float16* __restrict__ xh, const _Float16* __restrict__ wh,
    const float* __restrict__ bias,
    _Float16* __restrict__ Qh, _Float16* __restrict__ Kh, _Float16* __restrict__ Vt)
{
  const int wave = threadIdx.x >> 5, lane = threadIdx.x & 31;
  const int lm = lane & 15, lh = lane >> 4;
  const int ntl = blockIdx.x;                   // 0..39 (16-row tile within batch)
  const int b   = blockIdx.y;                   // 0..63
  const size_t rowA = (size_t)b * NPAD + ntl * 16 + lm;

  AF a[4];
  #pragma unroll
  for (int kc = 0; kc < 4; ++kc)
    #pragma unroll
    for (int j = 0; j < 8; ++j)
      a[kc].u[j] = *(const unsigned*)(xh + rowA * C__ + kc * 32 + kbase_of(j, lh));

  for (int t = 0; t < 3; ++t) {
    const int nt  = wave * 3 + t;               // 0..23 column tiles of 384
    const int col = nt * 16 + lm;
    v8f acc = {};
    #pragma unroll
    for (int kc = 0; kc < 4; ++kc) {
      v16h bf = *(const v16h*)(wh + (size_t)col * C__ + kc * 32 + lh * 16);
      acc = __builtin_amdgcn_wmma_f32_16x16x32_f16(false, a[kc].v, false, bf,
                                                   (short)0, acc, false, false);
    }
    const int p = col >> 7, hd = col & 127, hh = hd >> 5, d = hd & 31;
    const float bb = bias[col];
    const size_t bhh = (size_t)b * H__ + hh;
    const int n0 = ntl * 16 + lh * 8;           // C-frag rows for this lane
    if (p == 2) {                               // V transposed: stride 1 in n
      _Float16* dst = Vt + (bhh * D__ + d) * NPAD + n0;
      #pragma unroll
      for (int r = 0; r < 8; ++r)
        dst[r] = (_Float16)(acc[r] + bb);
    } else {                                    // Q/K row-major: stride D__ in n
      const float mulv = (p == 0) ? 0.17677669529663687f : 1.0f;
      _Float16* dst = ((p == 0) ? Qh : Kh) + (bhh * NPAD + n0) * D__ + d;
      #pragma unroll
      for (int r = 0; r < 8; ++r)
        dst[r * D__] = (_Float16)((acc[r] + bb) * mulv);
    }
  }
}

// ---------------- Kernel 2: biasHNN[h][q][k] = table[rel_index[q,k], h]
__global__ void bias_kernel(const float* __restrict__ tbl,
                            const int* __restrict__ ridx,
                            float* __restrict__ biasHNN)
{
  int idx = blockIdx.x * blockDim.x + threadIdx.x;
  if (idx >= N__ * N__) return;
  int ri = ridx[idx];
  #pragma unroll
  for (int h = 0; h < H__; ++h)
    biasHNN[(size_t)h * (N__ * N__) + idx] = tbl[ri * H__ + h];
}

// ---------------- Kernel 3: flash attention, one wave per (b,h,qtile)
__global__ __launch_bounds__(256) void attn_kernel(
    const _Float16* __restrict__ Qh, const _Float16* __restrict__ Kh,
    const _Float16* __restrict__ Vt, const float* __restrict__ biasHNN,
    const float* __restrict__ mask, _Float16* __restrict__ Z)
{
  __shared__ __align__(16) _Float16 pSh[8 * 512];   // per-wave 16x32 P staging
  const int wave = threadIdx.x >> 5, lane = threadIdx.x & 31;
  const int lm = lane & 15, lh = lane >> 4;
  const int bh = blockIdx.x / 5, qg = blockIdx.x % 5;
  const int b = bh >> 2, h = bh & 3, wdw = b & (NW__ - 1);
  const int qt = wave * 5 + qg;                     // 0..39
  _Float16* ps = pSh + wave * 512;

  const _Float16* Qb = Qh + (size_t)bh * NPAD * D__;
  const _Float16* Kb = Kh + (size_t)bh * NPAD * D__;
  const _Float16* Vb = Vt + (size_t)bh * D__ * NPAD;
  const float* biasB = biasHNN + (size_t)h * (N__ * N__);
  const float* maskB = mask + (size_t)wdw * (N__ * N__);

  AF aq;                                            // Q A-fragment (loaded once)
  {
    const int qrow = qt * 16 + lm;
    #pragma unroll
    for (int j = 0; j < 8; ++j)
      aq.u[j] = *(const unsigned*)(Qb + (size_t)qrow * D__ + kbase_of(j, lh));
  }

  float mrow[8], lrow[8];
  v8f o0 = {}, o1 = {};
  #pragma unroll
  for (int r = 0; r < 8; ++r) { mrow[r] = -3.0e38f; lrow[r] = 0.0f; }
  const int qbase = qt * 16 + lh * 8;

  for (int kc = 0; kc < 20; ++kc) {                 // 32 keys per step
    const int k0 = kc * 32;
    v16h bk0 = *(const v16h*)(Kb + (size_t)(k0 + lm)      * D__ + lh * 16);
    v16h bk1 = *(const v16h*)(Kb + (size_t)(k0 + 16 + lm) * D__ + lh * 16);
    v8f s0 = {}, s1 = {};
    s0 = __builtin_amdgcn_wmma_f32_16x16x32_f16(false, aq.v, false, bk0, (short)0, s0, false, false);
    s1 = __builtin_amdgcn_wmma_f32_16x16x32_f16(false, aq.v, false, bk1, (short)0, s1, false, false);

    const int kcol0 = k0 + lm, kcol1 = k0 + 16 + lm;
    #pragma unroll
    for (int r = 0; r < 8; ++r) {
      const int q = qbase + r;
      const int base = q * N__ + kcol0;
      float v0 = s0[r], v1 = s1[r];
      if (q < N__ && kcol0 < N__) v0 += biasB[base] + maskB[base];
      else v0 = -1.0e30f;
      if (q < N__ && kcol1 < N__) v1 += biasB[base + 16] + maskB[base + 16];
      else v1 = -1.0e30f;

      float mv = fmaxf(v0, v1);
      #pragma unroll
      for (int s = 1; s <= 8; s <<= 1) mv = fmaxf(mv, __shfl_xor(mv, s, 32));
      float mnew = fmaxf(mrow[r], mv);
      float p0 = __expf(v0 - mnew), p1 = __expf(v1 - mnew);
      float psum = p0 + p1;
      #pragma unroll
      for (int s = 1; s <= 8; s <<= 1) psum += __shfl_xor(psum, s, 32);
      float corr = __expf(mrow[r] - mnew);
      lrow[r] = lrow[r] * corr + psum;
      mrow[r] = mnew;
      o0[r] *= corr; o1[r] *= corr;

      const int m = lh * 8 + r;                     // stage P (C-layout -> LDS)
      ps[m * 32 + lm]      = (_Float16)p0;
      ps[m * 32 + 16 + lm] = (_Float16)p1;
    }
    asm volatile("s_wait_dscnt 0" ::: "memory");
    AF ap;                                          // P A-fragment from LDS
    #pragma unroll
    for (int j = 0; j < 8; ++j)
      ap.u[j] = *(const unsigned*)(ps + lm * 32 + kbase_of(j, lh));

    v16h bv0 = *(const v16h*)(Vb + (size_t)lm        * NPAD + k0 + lh * 16);
    v16h bv1 = *(const v16h*)(Vb + (size_t)(16 + lm) * NPAD + k0 + lh * 16);
    o0 = __builtin_amdgcn_wmma_f32_16x16x32_f16(false, ap.v, false, bv0, (short)0, o0, false, false);
    o1 = __builtin_amdgcn_wmma_f32_16x16x32_f16(false, ap.v, false, bv1, (short)0, o1, false, false);
  }

  #pragma unroll
  for (int r = 0; r < 8; ++r) {
    const int q = qbase + r;
    if (q < N__) {
      float inv = 1.0f / lrow[r];
      size_t zrow = ((size_t)b * N__ + q) * C__ + h * D__;
      Z[zrow + lm]      = (_Float16)(o0[r] * inv);
      Z[zrow + 16 + lm] = (_Float16)(o1[r] * inv);
    }
  }
}

// ---------------- Kernel 4: out = Z @ proj_w.T + proj_b  (f32 out)
__global__ __launch_bounds__(256) void proj_kernel(
    const _Float16* __restrict__ Z, const _Float16* __restrict__ wh,
    const float* __restrict__ bias, float* __restrict__ out)
{
  const int wave = threadIdx.x >> 5, lane = threadIdx.x & 31;
  const int lm = lane & 15, lh = lane >> 4;
  const int mt = blockIdx.x, nt = wave;            // 8 waves -> 8 col tiles
  const int rowA = mt * 16 + lm;
  const int col  = nt * 16 + lm;
  v8f acc = {};
  #pragma unroll
  for (int kc = 0; kc < 4; ++kc) {
    AF a;
    #pragma unroll
    for (int j = 0; j < 8; ++j)
      a.u[j] = *(const unsigned*)(Z + (size_t)rowA * C__ + kc * 32 + kbase_of(j, lh));
    v16h bf = *(const v16h*)(wh + (size_t)col * C__ + kc * 32 + lh * 16);
    acc = __builtin_amdgcn_wmma_f32_16x16x32_f16(false, a.v, false, bf,
                                                 (short)0, acc, false, false);
  }
  const float bb = bias[col];
  #pragma unroll
  for (int r = 0; r < 8; ++r) {
    int row = mt * 16 + lh * 8 + r;
    out[(size_t)row * C__ + col] = acc[r] + bb;
  }
}

extern "C" void kernel_launch(void* const* d_in, const int* in_sizes, int n_in,
                              void* d_out, int out_size, void* d_ws, size_t ws_size,
                              hipStream_t stream) {
  const float* x      = (const float*)d_in[0];
  const float* mask   = (const float*)d_in[1];
  const float* qkv_w  = (const float*)d_in[2];
  const float* qkv_b  = (const float*)d_in[3];
  const float* proj_w = (const float*)d_in[4];
  const float* proj_b = (const float*)d_in[5];
  const float* tbl    = (const float*)d_in[6];
  const int*   ridx   = (const int*)d_in[7];
  char* ws = (char*)d_ws;
  _Float16* Qh  = (_Float16*)(ws + QOFF);
  _Float16* Kh  = (_Float16*)(ws + KOFF);
  _Float16* Vt  = (_Float16*)(ws + VTOFF);
  _Float16* Z   = (_Float16*)(ws + ZOFF);
  float* biasHNN = (float*)(ws + BOFF);
  _Float16* xh  = (_Float16*)(ws + XHOFF);
  _Float16* wqh = (_Float16*)(ws + WQOFF);
  _Float16* wph = (_Float16*)(ws + WPOFF);

  // only xh pad rows are never rewritten -> zero just that region
  hipMemsetAsync(ws + XHOFF, 0, XHBYTES, stream);

  // f32 -> f16 pre-passes
  cvtx_kernel<<<dim3((N__ * (C__ / 4) + 255) / 256, B__), 256, 0, stream>>>(x, xh);
  cvt4_kernel<<<(384 * C__ / 4 + 255) / 256, 256, 0, stream>>>(qkv_w, wqh, 384 * C__ / 4);
  cvt4_kernel<<<(C__ * C__ / 4 + 255) / 256, 256, 0, stream>>>(proj_w, wph, C__ * C__ / 4);

  qkv_kernel <<<dim3(NPAD / 16, B__), 256, 0, stream>>>(xh, wqh, qkv_b, Qh, Kh, Vt);
  bias_kernel<<<(N__ * N__ + 255) / 256, 256, 0, stream>>>(tbl, ridx, biasHNN);
  attn_kernel<<<B__ * H__ * 5, 256, 0, stream>>>(Qh, Kh, Vt, biasHNN, mask, Z);
  proj_kernel<<<2500, 256, 0, stream>>>(Z, wph, proj_b, (float*)d_out);
}
